// LinearPlusSwishGatePos_24189255811504
// MI455X (gfx1250) — matrix operand order repaired and checked by reading, not compiled, not run on hardware
//
#include <hip/hip_runtime.h>
#include <hip/hip_bf16.h>

// ---------------------------------------------------------------------------
// LinearPlusSwishGatePos for MI455X (gfx1250, wave32, WMMA bf16 + async LDS).
//
// Roofline: 459 KFLOP / 5.1 KB per node -> memory bound (~22us floor at
// 23.3 TB/s) IF the GEMMs run on v_wmma. Weights (384KB bf16 packed) stay
// L2/WGP$-resident; HBM traffic is node data only (read exactly once).
// Phase A stages xv via GLOBAL_LOAD_ASYNC_TO_LDS_B128 (ASYNCcnt).
// ---------------------------------------------------------------------------

#define MUL_S   256
#define MUL_V   128
#define DIM_H   640          // MUL_S + 3*MUL_V
#define ROW_F   644          // DIM_H + 4

#define INV_SQRT3f 0.57735026918962576451f
#define C0f        0.05103103630798287f   // 1/sqrt(384)

// packed bf16 weight offsets (in 16-bit elements) inside d_ws
#define OFF_SS   0        // 256x256 : 8 ktiles x 16 ntiles
#define OFF_VV   65536    // 128x256 : 4 x 16
#define OFF_SSG  98304    // 256x128 : 8 x 8
#define OFF_VVG  131072   // 128x128 : 4 x 8
#define OFF_SV   147456   // 256x128 : 8 x 8
#define OFF_VS   180224   // 128x128 : 4 x 8

typedef __attribute__((ext_vector_type(16))) __bf16 v16bf;
typedef __attribute__((ext_vector_type(8)))  float  v8f;

struct Frag { union { v16bf b; __bf16 e[16]; unsigned short s[16]; }; };

static __device__ __forceinline__ unsigned short f2bf_rne(float f) {
    unsigned u = __float_as_uint(f);
    u += 0x7FFFu + ((u >> 16) & 1u);
    return (unsigned short)(u >> 16);
}

// ---------------------------------------------------------------------------
// Weight pre-pack: f32 [K x N] row-major -> bf16 WMMA-B tiles (32x16 = 512
// shorts, tile index = tn*ntk + tk). Lane l owns 16 shorts at l*16:
//   n = tn*16 + (l & 15), k = tk*32 + (l>=16 ? 8 : 0) + e + (e>=8 ? 8 : 0)
// so the main kernel reads each B fragment as one coalesced 32B/lane load.
// ---------------------------------------------------------------------------
__global__ void pack_w_kernel(const float* __restrict__ src,
                              unsigned short* __restrict__ dst,
                              int K, int N, int total) {
    int tid = blockIdx.x * 256 + threadIdx.x;
    if (tid >= total) return;
    int tile   = tid >> 9;
    int within = tid & 511;
    int l = within >> 4;
    int e = within & 15;
    int ntk = K >> 5;
    int tn = tile / ntk;
    int tk = tile - tn * ntk;
    int k = tk * 32 + ((l & 16) ? 8 : 0) + e + ((e & 8) ? 8 : 0);
    int n = tn * 16 + (l & 15);
    dst[tid] = f2bf_rne(src[k * N + n]);
}

// ---------------------------------------------------------------------------
// Main fused kernel. Block = 128 threads = 4 waves; each wave owns a 16-node
// M-tile (block = 64 nodes). Per-wave LDS: xv stage [16][388]f32, dot[16][132],
// {ps,pv} [16][4] = 33536 B; 4 waves = 134144 B dynamic LDS (2 blocks/WGP).
// ---------------------------------------------------------------------------
#define WAVE_F  8384                  // floats of LDS per wave
#define XV_STR  388                   // 384 + pad
#define DOT_STR 132                   // 128 + pad

__global__ __launch_bounds__(128)
void fused_lpsgp_kernel(const float* __restrict__ in,
                        const unsigned short* __restrict__ wp,
                        const float* __restrict__ bias,
                        float* __restrict__ out,
                        int n_nodes) {
    extern __shared__ float smem[];
    const int tid  = threadIdx.x;
    const int wave = tid >> 5;
    const int lane = tid & 31;
    const int half = lane >> 4;       // K-half of WMMA lane layout
    const int mrow = lane & 15;       // row of M-tile / column of N-tile

    float* sw    = smem + wave * WAVE_F;
    float* xv_s  = sw;                              // [16][XV_STR]
    float* dot_s = sw + 16 * XV_STR;                // [16][DOT_STR]
    float* pp    = sw + 16 * XV_STR + 16 * DOT_STR; // [16][4] = {ps,pv0..2}

    const long long base = (long long)blockIdx.x * 64 + wave * 16;
    const bool full = (base + 16 <= (long long)n_nodes);   // uniform per wave

    // ---------------- Phase A: async-stage xv + ps/pv, compute dot ----------
    {
        const unsigned goff = (unsigned)(lane * 16);
        for (int r = 0; r < 16; ++r) {
            long long node = base + r;
            if (full || node < n_nodes) {
                // 16 rows x 1536B each: 3 async 512B wave-transfers per row.
                unsigned ldsoff =
                    (unsigned)(unsigned long long)(void*)(xv_s + r * XV_STR)
                    + (unsigned)(lane * 16);
                const float* rowp = in + (size_t)node * ROW_F + MUL_S;
                asm volatile(
                    "global_load_async_to_lds_b128 %0, %1, %2 offset:0\n\t"
                    "global_load_async_to_lds_b128 %0, %1, %2 offset:512\n\t"
                    "global_load_async_to_lds_b128 %0, %1, %2 offset:1024"
                    :: "v"(ldsoff), "v"(goff), "s"(rowp) : "memory");
                if (lane == 0)
                    *(float4*)(pp + r * 4) =
                        *(const float4*)(in + (size_t)node * ROW_F + DIM_H);
            } else {
                float4 z = make_float4(0.f, 0.f, 0.f, 0.f);
                float4* dst = (float4*)(xv_s + r * XV_STR);
                #pragma unroll
                for (int i = lane; i < 96; i += 32) dst[i] = z;
                if (lane == 0) *(float4*)(pp + r * 4) = z;
            }
        }
        asm volatile("s_wait_asynccnt 0x0" ::: "memory");
    }
    __syncthreads();

    {   // dot[m][u] = (xv[m][u][:] . pv[m]) * inv_sqrt3   (u pairs, b64 LDS)
        int m = lane >> 1, h = lane & 1;
        float pv0 = pp[m * 4 + 1], pv1 = pp[m * 4 + 2], pv2 = pp[m * 4 + 3];
        const float* xrow = xv_s + m * XV_STR;
        float*       drow = dot_s + m * DOT_STR;
        for (int u = h * 64; u < h * 64 + 64; u += 2) {
            float2 ab = *(const float2*)(xrow + 3 * u);
            float2 cd = *(const float2*)(xrow + 3 * u + 2);
            float2 ef = *(const float2*)(xrow + 3 * u + 4);
            float d0 = (ab.x * pv0 + ab.y * pv1 + cd.x * pv2) * INV_SQRT3f;
            float d1 = (cd.y * pv0 + ef.x * pv1 + ef.y * pv2) * INV_SQRT3f;
            *(float2*)(drow + u) = make_float2(d0, d1);
        }
    }
    __syncthreads();

    // ---------------- Phase B prologue: build A fragments once --------------
    const long long mynode = base + mrow;
    const bool mvalid = full || (mynode < n_nodes);
    const float ps_m = pp[mrow * 4 + 0];
    const float* myrow = in + (size_t)mynode * ROW_F;

    Frag a_xs[8], a_s0[8], a_dot[4], a_xv[3][4];

    #pragma unroll
    for (int kk = 0; kk < 8; ++kk) {          // xs / s0 : K = 256 (global)
        float t[16];
        int cb = kk * 32 + half * 8;
        if (mvalid) {
            *(float4*)(t +  0) = *(const float4*)(myrow + cb);
            *(float4*)(t +  4) = *(const float4*)(myrow + cb + 4);
            *(float4*)(t +  8) = *(const float4*)(myrow + cb + 16);
            *(float4*)(t + 12) = *(const float4*)(myrow + cb + 20);
        } else {
            #pragma unroll
            for (int i = 0; i < 16; ++i) t[i] = 0.f;
        }
        #pragma unroll
        for (int i = 0; i < 16; ++i) {
            a_xs[kk].e[i] = (__bf16)t[i];
            a_s0[kk].e[i] = (__bf16)(t[i] * ps_m);
        }
    }
    #pragma unroll
    for (int kk = 0; kk < 4; ++kk) {          // dot : K = 128 (LDS)
        const float* drow = dot_s + mrow * DOT_STR + kk * 32 + half * 8;
        float t[16];
        *(float4*)(t +  0) = *(const float4*)(drow);
        *(float4*)(t +  4) = *(const float4*)(drow + 4);
        *(float4*)(t +  8) = *(const float4*)(drow + 16);
        *(float4*)(t + 12) = *(const float4*)(drow + 20);
        #pragma unroll
        for (int i = 0; i < 16; ++i) a_dot[kk].e[i] = (__bf16)t[i];
    }
    #pragma unroll
    for (int k = 0; k < 3; ++k)               // xv_k : K = 128, stride-3 (LDS)
        #pragma unroll
        for (int kk = 0; kk < 4; ++kk) {
            const float* xrow = xv_s + mrow * XV_STR + k;
            int u0 = kk * 32 + half * 8;
            #pragma unroll
            for (int i = 0; i < 8; ++i) {
                a_xv[k][kk].e[i]     = (__bf16)xrow[3 * (u0 + i)];
                a_xv[k][kk].e[8 + i] = (__bf16)xrow[3 * (u0 + 16 + i)];
            }
        }

    const unsigned short* wl = wp + lane * 16;   // per-lane B base

    // ---------------- scal = silu(C0*(s0@W_ss + dot@W_vv) + b) --------------
    for (int j = 0; j < 16; ++j) {
        v8f acc = {};
        #pragma unroll
        for (int kk = 0; kk < 8; ++kk) {
            v16bf bt = *(const v16bf*)(wl + OFF_SS + (j * 8 + kk) * 512);
            acc = __builtin_amdgcn_wmma_f32_16x16x32_bf16(
                      false, a_s0[kk].b, false, bt, (short)0, acc, false, false);
        }
        #pragma unroll
        for (int kk = 0; kk < 4; ++kk) {
            v16bf bt = *(const v16bf*)(wl + OFF_VV + (j * 4 + kk) * 512);
            acc = __builtin_amdgcn_wmma_f32_16x16x32_bf16(
                      false, a_dot[kk].b, false, bt, (short)0, acc, false, false);
        }
        const int col = j * 16 + mrow;
        const float bc = bias[col];
        if (full) {
            #pragma unroll
            for (int r = 0; r < 8; ++r) {
                long long node = base + half * 8 + r;
                float v = C0f * acc[r] + bc;
                out[(size_t)node * DIM_H + col] = v / (1.f + __expf(-v));
            }
        } else {
            #pragma unroll
            for (int r = 0; r < 8; ++r) {
                long long node = base + half * 8 + r;
                if (node < n_nodes) {
                    float v = C0f * acc[r] + bc;
                    out[(size_t)node * DIM_H + col] = v / (1.f + __expf(-v));
                }
            }
        }
    }

    // ------- gate/vec: vec = C0*((xs@W_sv)*pv + (xv_k@W_vs)*ps) * sig(gate) -
    for (int j = 0; j < 8; ++j) {
        v8f gacc = {}, Gacc = {}, H0 = {}, H1 = {}, H2 = {};
        #pragma unroll
        for (int kk = 0; kk < 8; ++kk) {
            v16bf bg = *(const v16bf*)(wl + OFF_SSG + (j * 8 + kk) * 512);
            gacc = __builtin_amdgcn_wmma_f32_16x16x32_bf16(
                       false, a_s0[kk].b, false, bg, (short)0, gacc, false, false);
            v16bf bs = *(const v16bf*)(wl + OFF_SV + (j * 8 + kk) * 512);
            Gacc = __builtin_amdgcn_wmma_f32_16x16x32_bf16(
                       false, a_xs[kk].b, false, bs, (short)0, Gacc, false, false);
        }
        #pragma unroll
        for (int kk = 0; kk < 4; ++kk) {
            v16bf bg = *(const v16bf*)(wl + OFF_VVG + (j * 4 + kk) * 512);
            gacc = __builtin_amdgcn_wmma_f32_16x16x32_bf16(
                       false, a_dot[kk].b, false, bg, (short)0, gacc, false, false);
            v16bf bv = *(const v16bf*)(wl + OFF_VS + (j * 4 + kk) * 512);
            H0 = __builtin_amdgcn_wmma_f32_16x16x32_bf16(
                     false, a_xv[0][kk].b, false, bv, (short)0, H0, false, false);
            H1 = __builtin_amdgcn_wmma_f32_16x16x32_bf16(
                     false, a_xv[1][kk].b, false, bv, (short)0, H1, false, false);
            H2 = __builtin_amdgcn_wmma_f32_16x16x32_bf16(
                     false, a_xv[2][kk].b, false, bv, (short)0, H2, false, false);
        }
        const int w = j * 16 + mrow;
        #pragma unroll
        for (int r = 0; r < 8; ++r) {
            int row = half * 8 + r;
            long long node = base + row;
            if (full || node < n_nodes) {
                float g   = 1.f / (1.f + __expf(-C0f * gacc[r]));
                float psr = pp[row * 4 + 0];
                float p0  = pp[row * 4 + 1];
                float p1  = pp[row * 4 + 2];
                float p2  = pp[row * 4 + 3];
                float Gv  = Gacc[r];
                size_t o = (size_t)node * DIM_H + MUL_S + (size_t)w * 3;
                out[o + 0] = C0f * (Gv * p0 + H0[r] * psr) * g;
                out[o + 1] = C0f * (Gv * p1 + H1[r] * psr) * g;
                out[o + 2] = C0f * (Gv * p2 + H2[r] * psr) * g;
            }
        }
    }
}

// ---------------------------------------------------------------------------
extern "C" void kernel_launch(void* const* d_in, const int* in_sizes, int n_in,
                              void* d_out, int out_size, void* d_ws, size_t ws_size,
                              hipStream_t stream) {
    const float* data_in = (const float*)d_in[0];
    const float* W_ss    = (const float*)d_in[1];
    const float* W_vv    = (const float*)d_in[2];
    const float* W_ss_g  = (const float*)d_in[3];
    const float* W_vv_g  = (const float*)d_in[4];
    const float* W_sv    = (const float*)d_in[5];
    const float* W_vs    = (const float*)d_in[6];
    const float* b       = (const float*)d_in[7];
    float* out = (float*)d_out;
    unsigned short* wpack = (unsigned short*)d_ws;

    const int n_nodes = in_sizes[0] / ROW_F;

    // pack weights -> bf16 WMMA-B tiles (L2-resident, 384 KB total)
    pack_w_kernel<<<(65536 + 255) / 256, 256, 0, stream>>>(W_ss,   wpack + OFF_SS,  256, 256, 65536);
    pack_w_kernel<<<(32768 + 255) / 256, 256, 0, stream>>>(W_vv,   wpack + OFF_VV,  128, 256, 32768);
    pack_w_kernel<<<(32768 + 255) / 256, 256, 0, stream>>>(W_ss_g, wpack + OFF_SSG, 256, 128, 32768);
    pack_w_kernel<<<(16384 + 255) / 256, 256, 0, stream>>>(W_vv_g, wpack + OFF_VVG, 128, 128, 16384);
    pack_w_kernel<<<(32768 + 255) / 256, 256, 0, stream>>>(W_sv,   wpack + OFF_SV,  256, 128, 32768);
    pack_w_kernel<<<(16384 + 255) / 256, 256, 0, stream>>>(W_vs,   wpack + OFF_VS,  128, 128, 16384);

    const int blocks = (n_nodes + 63) / 64;
    const size_t lds_bytes = (size_t)4 * WAVE_F * sizeof(float); // 134144 B
    fused_lpsgp_kernel<<<blocks, 128, lds_bytes, stream>>>(data_in, wpack, b, out, n_nodes);
}
